// MultiHeadSelfAttention_82575041233466
// MI455X (gfx1250) — matrix operand order repaired and checked by reading, not compiled, not run on hardware
//
#include <hip/hip_runtime.h>

#define BB 16
#define SS 1024
#define DD 768
#define HH 12
#define DH 64

typedef __attribute__((ext_vector_type(16))) _Float16 v16h;
typedef __attribute__((ext_vector_type(8)))  _Float16 v8h;
typedef __attribute__((ext_vector_type(8)))  float    v8f;
typedef int v4i __attribute__((vector_size(16)));

#define AS1 __attribute__((address_space(1)))
#define AS3 __attribute__((address_space(3)))

#if __has_builtin(__builtin_amdgcn_global_load_async_to_lds_b128)
#define HAVE_ASYNC 1
#else
#define HAVE_ASYNC 0
#endif

// Copy 16 bytes global -> LDS. Async (ASYNCcnt) when the gfx1250 builtin exists.
static __device__ inline void copy16(const _Float16* g, _Float16* l) {
#if HAVE_ASYNC
  __builtin_amdgcn_global_load_async_to_lds_b128((AS1 v4i*)g, (AS3 v4i*)l, 0, 0);
#else
  *(v8h*)l = *(const v8h*)g;
#endif
}

static __device__ inline void wait_async() {
#if HAVE_ASYNC
#if __has_builtin(__builtin_amdgcn_s_wait_asynccnt)
  __builtin_amdgcn_s_wait_asynccnt(0);
#else
  asm volatile("s_wait_asynccnt 0x0" ::: "memory");
#endif
#endif
}

static __device__ inline v8f wmma_f16(v16h a, v16h b, v8f c) {
  return __builtin_amdgcn_wmma_f32_16x16x32_f16(false, a, false, b, (short)0, c, false, false);
}

static __device__ inline v16h cat8(v8h lo, v8h hi) {
  return __builtin_shufflevector(lo, hi, 0,1,2,3,4,5,6,7,8,9,10,11,12,13,14,15);
}

// A-fragment (16x32 f16, M x K) from a row-major f16 matrix.
// lanes: M=lane%16, elems 0..15 = K{hi*16..hi*16+15} (contiguous, matches load_bt)
static __device__ inline v16h load_a(const _Float16* row_base, int ld, int kb) {
  int lane = threadIdx.x & 31;
  const _Float16* p = row_base + (size_t)(lane & 15) * ld + kb * 32 + (lane >> 4) * 16;
  v8h lo = *(const v8h*)p;
  v8h hi = *(const v8h*)(p + 8);
  return cat8(lo, hi);
}

// B-fragment (32x16 f16, K x N) where source is stored as src[n][k] row-major:
// lane = n + 16*khalf, elems = K khalf*16+e -> one contiguous 16-half chunk/lane.
static __device__ inline v16h load_bt(const _Float16* src, size_t ld) {
  int lane = threadIdx.x & 31;
  return *(const v16h*)(src + (size_t)(lane & 15) * ld + (size_t)(lane >> 4) * 16);
}

// ---------------- Kernel 0: W_last f32 -> transposed f16 [out][in] ----------
__global__ __launch_bounds__(256) void wconv_kernel(const float* __restrict__ W,
                                                    _Float16* __restrict__ wT16) {
  int t = blockIdx.x * 256 + threadIdx.x;
  if (t < DD * DD) {
    int i = t / DD, j = t % DD;                 // W[i][j] (in, out)
    wT16[(size_t)j * DD + i] = (_Float16)W[t];  // wT[out][in]
  }
}

// ---------------- Kernel 1: QKV projection -------------------------------
// grid (S/64, H, B), 256 threads. q16,k16: [B*H][S][DH] f16 row-major.
// vT16: [B*H][DH][S] f16 (transposed so the P*V B-fragment is contiguous).
__global__ __launch_bounds__(256) void qkv_kernel(
    const float* __restrict__ seq,
    const float* __restrict__ Wq, const float* __restrict__ bq,
    const float* __restrict__ Wk, const float* __restrict__ bk,
    const float* __restrict__ Wv, const float* __restrict__ bv,
    _Float16* __restrict__ q16, _Float16* __restrict__ k16,
    _Float16* __restrict__ vT16) {
  __shared__ _Float16 xs[64 * 64];
  __shared__ _Float16 wT[3][64 * 64];
  int tid = threadIdx.x;
  int st = blockIdx.x, h = blockIdx.y, b = blockIdx.z;
  int s0 = st * 64;
  for (int idx = tid; idx < 4096; idx += 256) {
    int r = idx >> 6, c = idx & 63;
    xs[idx] = (_Float16)seq[((size_t)b * SS + s0 + r) * DD + h * DH + c];
    // W[h][in][out] row-major -> wT[out][in] for contiguous B-frags
    wT[0][c * 64 + r] = (_Float16)Wq[(size_t)h * 4096 + idx];
    wT[1][c * 64 + r] = (_Float16)Wk[(size_t)h * 4096 + idx];
    wT[2][c * 64 + r] = (_Float16)Wv[(size_t)h * 4096 + idx];
  }
  __syncthreads();

  int wid = tid >> 5, lane = tid & 31, n15 = lane & 15, hi = lane >> 4;
  size_t bh = (size_t)b * HH + h;
  for (int job = wid; job < 48; job += 8) {
    int mat = job >> 4, tile = job & 15, mt = tile >> 2, nt = tile & 3;
    v16h a0 = load_a(xs + mt * 16 * 64, 64, 0);
    v16h a1 = load_a(xs + mt * 16 * 64, 64, 1);
    v16h b0 = load_bt(wT[mat] + (size_t)nt * 16 * 64, 64);
    v16h b1 = load_bt(wT[mat] + (size_t)nt * 16 * 64 + 32, 64);
    v8f acc = {};
    acc = wmma_f16(a0, b0, acc);
    acc = wmma_f16(a1, b1, acc);
    int col = nt * 16 + n15;
    const float* bias = (mat == 0) ? bq : (mat == 1) ? bk : bv;
    float bv_ = bias[h * DH + col];
    if (mat == 2) {
      v8h pk;
#pragma unroll
      for (int r = 0; r < 8; ++r) pk[r] = (_Float16)(acc[r] + bv_);
      *(v8h*)(vT16 + ((bh * DH + col) * SS + s0 + mt * 16 + hi * 8)) = pk;
    } else {
      _Float16* dst = (mat == 0 ? q16 : k16) +
                      (bh * SS + s0 + mt * 16 + hi * 8) * DH + col;
#pragma unroll
      for (int r = 0; r < 8; ++r) dst[(size_t)r * DH] = (_Float16)(acc[r] + bv_);
    }
  }
}

// ---------------- Kernel 2: flash attention ------------------------------
// grid (S/128, H, B), 256 threads (8 waves x 16 query rows). K/V tiles are
// double-buffered in LDS via async global->LDS copies and shared by all waves
// (8x less L2 traffic than per-wave streaming).
__global__ __launch_bounds__(256) void attn_kernel(
    const _Float16* __restrict__ q16, const _Float16* __restrict__ k16,
    const _Float16* __restrict__ vT16, _Float16* __restrict__ att16) {
  __shared__ _Float16 kbuf[2][32 * 64];   // [key][dh]
  __shared__ _Float16 vbuf[2][64 * 32];   // [dh][key]
  __shared__ _Float16 plds[8][16 * 32];   // wave-private P re-layout
  int tid = threadIdx.x, wid = tid >> 5, lane = tid & 31;
  int n15 = lane & 15, hi = lane >> 4;
  int qt = blockIdx.x, h = blockIdx.y, b = blockIdx.z;
  int s0 = qt * 128 + wid * 16;
  size_t bh = (size_t)b * HH + h;

  const _Float16* qbase = q16 + (bh * SS + s0) * DH;
  v16h aq0 = load_a(qbase, DH, 0);
  v16h aq1 = load_a(qbase, DH, 1);

  float m_r[8], l_r[8];
  v8f o0 = {}, o1 = {}, o2 = {}, o3 = {};
#pragma unroll
  for (int r = 0; r < 8; ++r) { m_r[r] = -1e30f; l_r[r] = 0.f; }
  _Float16* pw = plds[wid];
  const float scale2 = 0.125f * 1.44269504088896340736f;  // (1/sqrt(64))*log2(e)

  // cooperative copy assignment: 256 thr x (16B K-chunk + 16B V-chunk) = 8KB/tile
  int kc_row = tid >> 3, kc_off = (tid & 7) * 8;  // K: 32 rows x 8 chunks
  int vc_dh  = tid >> 2, vc_off = (tid & 3) * 8;  // V: 64 rows x 4 chunks

  auto issue_tile = [&](int t, int bufi) {
    copy16(k16 + (bh * SS + t + kc_row) * DH + kc_off,
           kbuf[bufi] + kc_row * 64 + kc_off);
    copy16(vT16 + (bh * DH + vc_dh) * SS + t + vc_off,
           vbuf[bufi] + vc_dh * 32 + vc_off);
  };
  issue_tile(0, 0);

  for (int t = 0; t < SS; t += 32) {
    int bufi = (t >> 5) & 1;
    wait_async();       // own async copies of tile t landed
    __syncthreads();    // everyone's copies landed; prior reads of other buf done
    if (t + 32 < SS) issue_tile(t + 32, bufi ^ 1);

    const _Float16* kb_ = kbuf[bufi];
    v16h bk00 = *(const v16h*)(kb_ + n15 * 64 + hi * 16);        // keys 0..15, dh 0..31
    v16h bk01 = *(const v16h*)(kb_ + n15 * 64 + 32 + hi * 16);   //             dh 32..63
    v16h bk10 = *(const v16h*)(kb_ + (16 + n15) * 64 + hi * 16); // keys 16..31
    v16h bk11 = *(const v16h*)(kb_ + (16 + n15) * 64 + 32 + hi * 16);
    v8f sc0 = {}, sc1 = {};
    sc0 = wmma_f16(aq0, bk00, sc0);
    sc0 = wmma_f16(aq1, bk01, sc0);
    sc1 = wmma_f16(aq0, bk10, sc1);
    sc1 = wmma_f16(aq1, bk11, sc1);

#pragma unroll
    for (int r = 0; r < 8; ++r) {
      float a = sc0[r] * scale2, c = sc1[r] * scale2;
      float mx = fmaxf(a, c);
#pragma unroll
      for (int d = 8; d >= 1; d >>= 1) mx = fmaxf(mx, __shfl_xor(mx, d, 16));
      float mn = fmaxf(m_r[r], mx);
      float p0 = exp2f(a - mn), p1 = exp2f(c - mn);
      float corr = exp2f(m_r[r] - mn);
      float rs = p0 + p1;
#pragma unroll
      for (int d = 8; d >= 1; d >>= 1) rs += __shfl_xor(rs, d, 16);
      l_r[r] = l_r[r] * corr + rs;
      m_r[r] = mn;
      o0[r] *= corr; o1[r] *= corr; o2[r] *= corr; o3[r] *= corr;
      int row = r + 8 * hi;
      pw[row * 32 + n15]      = (_Float16)p0;
      pw[row * 32 + 16 + n15] = (_Float16)p1;
    }
    // LDS is in-order per wave (DScnt): safe to reload our own region.
    v8h plo = *(const v8h*)(pw + n15 * 32 + hi * 16);
    v8h phi = *(const v8h*)(pw + n15 * 32 + hi * 16 + 8);
    v16h ap = cat8(plo, phi);

    const _Float16* vb_ = vbuf[bufi];
    o0 = wmma_f16(ap, *(const v16h*)(vb_ + (size_t)(0 * 16 + n15) * 32 + hi * 16), o0);
    o1 = wmma_f16(ap, *(const v16h*)(vb_ + (size_t)(1 * 16 + n15) * 32 + hi * 16), o1);
    o2 = wmma_f16(ap, *(const v16h*)(vb_ + (size_t)(2 * 16 + n15) * 32 + hi * 16), o2);
    o3 = wmma_f16(ap, *(const v16h*)(vb_ + (size_t)(3 * 16 + n15) * 32 + hi * 16), o3);
  }

#pragma unroll
  for (int r = 0; r < 8; ++r) {
    float inv = 1.0f / l_r[r];
    size_t row = (size_t)b * SS + s0 + r + 8 * hi;
    _Float16* dst = att16 + row * DD + h * DH + n15;
    dst[0]  = (_Float16)(o0[r] * inv);
    dst[16] = (_Float16)(o1[r] * inv);
    dst[32] = (_Float16)(o2[r] * inv);
    dst[48] = (_Float16)(o3[r] * inv);
  }
}

// ---------------- Kernel 3: final projection + bias ----------------------
// out[16384,768] = att16[16384,768] * W_last[768,768] + b. grid (128, 12), 256 thr.
__global__ __launch_bounds__(256) void proj_kernel(
    const _Float16* __restrict__ att16, const _Float16* __restrict__ wT16,
    const float* __restrict__ bias, float* __restrict__ out) {
  int tid = threadIdx.x, wid = tid >> 5, lane = tid & 31;
  int n15 = lane & 15, hi = lane >> 4;
  size_t row0 = (size_t)blockIdx.x * 128 + (size_t)wid * 16;
  int col0 = blockIdx.y * 64;
  v8f a0 = {}, a1 = {}, a2 = {}, a3 = {};
  const _Float16* abase = att16 + row0 * DD;
  for (int kb = 0; kb < DD / 32; ++kb) {
    v16h a = load_a(abase, DD, kb);
    const _Float16* wb = wT16 + (size_t)col0 * DD + kb * 32;
    a0 = wmma_f16(a, load_bt(wb, DD), a0);
    a1 = wmma_f16(a, load_bt(wb + (size_t)16 * DD, DD), a1);
    a2 = wmma_f16(a, load_bt(wb + (size_t)32 * DD, DD), a2);
    a3 = wmma_f16(a, load_bt(wb + (size_t)48 * DD, DD), a3);
  }
  float b0 = bias[col0 + n15], b1 = bias[col0 + 16 + n15];
  float b2 = bias[col0 + 32 + n15], b3 = bias[col0 + 48 + n15];
#pragma unroll
  for (int r = 0; r < 8; ++r) {
    float* dst = out + (row0 + r + 8 * hi) * DD + col0 + n15;
    dst[0]  = a0[r] + b0;
    dst[16] = a1[r] + b1;
    dst[32] = a2[r] + b2;
    dst[48] = a3[r] + b3;
  }
}

extern "C" void kernel_launch(void* const* d_in, const int* in_sizes, int n_in,
                              void* d_out, int out_size, void* d_ws, size_t ws_size,
                              hipStream_t stream) {
  const float* seq = (const float*)d_in[0];
  const float* Wq  = (const float*)d_in[1];
  const float* bq  = (const float*)d_in[2];
  const float* Wk  = (const float*)d_in[3];
  const float* bk  = (const float*)d_in[4];
  const float* Wv  = (const float*)d_in[5];
  const float* bv  = (const float*)d_in[6];
  const float* Wl  = (const float*)d_in[7];
  const float* bl  = (const float*)d_in[8];
  float* out = (float*)d_out;

  size_t nqkv = (size_t)BB * HH * SS * DH;   // 12,582,912 halves per tensor
  _Float16* q16   = (_Float16*)d_ws;
  _Float16* k16   = q16 + nqkv;
  _Float16* vT16  = k16 + nqkv;
  _Float16* att16 = vT16 + nqkv;
  _Float16* wT16  = att16 + (size_t)BB * SS * DD;

  wconv_kernel<<<(DD * DD + 255) / 256, 256, 0, stream>>>(Wl, wT16);
  qkv_kernel<<<dim3(SS / 64, HH, BB), 256, 0, stream>>>(seq, Wq, bq, Wk, bk, Wv, bv,
                                                        q16, k16, vT16);
  attn_kernel<<<dim3(SS / 128, HH, BB), 256, 0, stream>>>(q16, k16, vT16, att16);
  proj_kernel<<<dim3((BB * SS) / 128, DD / 64), 256, 0, stream>>>(att16, wT16, bl, out);
}
